// SelfAttention_74998718923523
// MI455X (gfx1250) — matrix operand order repaired and checked by reading, not compiled, not run on hardware
//
#include <hip/hip_runtime.h>

// Self-attention (SAGAN-style) for MI455X / gfx1250.
// Q = Wk*x+bk, K = Wq*x+bq, V = Wv*x+bv ; out = gamma * V softmax_i(K^T Q).
// All matmuls via v_wmma_f32_16x16x32_f16 (f16 in, f32 accumulate).
// Flash-attention tiling: no [N][N] score matrix (would be 256MB > L2).
// Workspace use (~10.2 MB): Wf swizzled weights + Qh/Kh [B][N][32] f16 + Vh [B][C][N] f16.

typedef __attribute__((ext_vector_type(16))) _Float16 v16h;
typedef __attribute__((ext_vector_type(8)))  _Float16 v8h;
typedef __attribute__((ext_vector_type(8)))  float    v8f;

#define B_  4
#define C_  256
#define CK_ 32
#define N_  4096

union F16Frag { v16h v; v8h h[2]; _Float16 e[16]; };

// ---------------------------------------------------------------------------
// Kernel 1: swizzle concatenated [Wq;Wk;Wv] (320x256 f32) into WMMA A-fragment
// order (f16): Wf[rt][step][lane][16], so each lane's A-frag is one 32B load.
// A 16x32 f16 layout: lane (h=lane/16, r=lane%16) elem j: vgpr v=j/2, s=j&1,
//   k = (v<4) ? h*8+2v+s : 16+h*8+2(v-4)+s
// ---------------------------------------------------------------------------
__global__ __launch_bounds__(256) void swizzle_w(const float* __restrict__ Wq,
                                                 const float* __restrict__ Wk,
                                                 const float* __restrict__ Wv,
                                                 _Float16* __restrict__ Wf) {
  int idx  = blockIdx.x * 256 + threadIdx.x;     // < 20*8*32*16 = 81920
  int j    = idx & 15;
  int lane = (idx >> 4) & 31;
  int step = (idx >> 9) & 7;
  int rt   = idx >> 12;                          // 0..19 (row tile of 16)
  int h = lane >> 4, r = lane & 15;
  int v = j >> 1,  s = j & 1;
  int kl = (v < 4) ? (h * 8 + 2 * v + s) : (16 + h * 8 + 2 * (v - 4) + s);
  int c = step * 32 + kl;
  int row = rt * 16 + r;                         // rows 0..31=Wq(->K), 32..63=Wk(->Q), 64..319=Wv(->V)
  float w;
  if (row < 32)      w = Wq[row * C_ + c];
  else if (row < 64) w = Wk[(row - 32) * C_ + c];
  else               w = Wv[(row - 64) * C_ + c];
  Wf[idx] = (_Float16)w;
}

// ---------------------------------------------------------------------------
// Kernel 2: projections via WMMA. One WG (8 waves) handles 64 positions:
// stage x[256c x 64n] -> LDS as [n][c] f16, then 80 16x16 output tiles
// (20 row-tiles x 4 n-tiles), K=256 in 8 steps of 32.
// B 32x16 f16 layout: lane holds col n=lane%16; elems j are k = h*8 + j (j<8), 16+h*8+j-8 (j>=8)
// (same order as A) -> two 16B LDS reads per fragment.
// ---------------------------------------------------------------------------
__global__ __launch_bounds__(256) void proj_qkv(const float* __restrict__ x,
                                                const _Float16* __restrict__ Wf,
                                                const float* __restrict__ bq,
                                                const float* __restrict__ bk,
                                                const float* __restrict__ bv,
                                                _Float16* __restrict__ Qh,
                                                _Float16* __restrict__ Kh,
                                                _Float16* __restrict__ Vh) {
  const int b  = blockIdx.y;
  const int n0 = blockIdx.x * 64;
  __shared__ _Float16 xT[64][272];               // [n][c], padded stride
  const int t = threadIdx.x;
  {
    const int nl = t & 63;                       // coalesced over n
    const int cq = t >> 6;                       // 4 channel quarters
    const float* xp = x + ((size_t)b * C_ + cq * 64) * N_ + n0 + nl;
#pragma unroll 4
    for (int i = 0; i < 64; ++i)
      xT[nl][cq * 64 + i] = (_Float16)xp[(size_t)i * N_];
  }
  __syncthreads();

  const int wv = t >> 5, lane = t & 31, h = lane >> 4, r = lane & 15;
  for (int k = 0; k < 10; ++k) {
    const int idx = wv + 8 * k;                  // 0..79
    const int rt = idx >> 2, ct = idx & 3;
    v8f acc;
#pragma unroll
    for (int j = 0; j < 8; ++j) {                // bias in accumulator init
      int grow = rt * 16 + 8 * h + j;
      acc[j] = (grow < 32) ? bq[grow] : (grow < 64) ? bk[grow - 32] : bv[grow - 64];
    }
#pragma unroll
    for (int step = 0; step < 8; ++step) {
      v16h a = *(const v16h*)(Wf + ((size_t)((rt * 8 + step) * 32 + lane)) * 16);
      F16Frag bf;
      const _Float16* bp = &xT[ct * 16 + r][step * 32 + h * 8];
      bf.h[0] = *(const v8h*)bp;
      bf.h[1] = *(const v8h*)(bp + 16);
      acc = __builtin_amdgcn_wmma_f32_16x16x32_f16(false, a, false, bf.v,
                                                   (short)0, acc, false, false);
    }
    const int n = n0 + ct * 16 + r;
    if (rt < 2) {                                // keys (f = Wq*x+bq), [n][32]
#pragma unroll
      for (int j = 0; j < 8; ++j)
        Kh[((size_t)b * N_ + n) * CK_ + rt * 16 + 8 * h + j] = (_Float16)acc[j];
    } else if (rt < 4) {                         // queries (g = Wk*x+bk), [n][32]
#pragma unroll
      for (int j = 0; j < 8; ++j)
        Qh[((size_t)b * N_ + n) * CK_ + (rt - 2) * 16 + 8 * h + j] = (_Float16)acc[j];
    } else {                                     // values (h = Wv*x+bv), [c][n]
#pragma unroll
      for (int j = 0; j < 8; ++j)
        Vh[((size_t)b * C_ + (rt - 4) * 16 + 8 * h + j) * N_ + n] = (_Float16)acc[j];
    }
  }
}

// ---------------------------------------------------------------------------
// Kernel 3: flash attention. WG = 8 waves; each wave owns 16 queries and the
// full 256-channel output (16 f32 accum tiles = 128 VGPRs). Per 32-key block:
// S = Q K^T (2 WMMA) -> online softmax in wave-private LDS -> P (f16) -> 16
// WMMA for O += P V. K/V are L2-resident (2.25 MB/batch).
// ---------------------------------------------------------------------------
__global__ __launch_bounds__(256) void flash_attn(const _Float16* __restrict__ Qh,
                                                  const _Float16* __restrict__ Kh,
                                                  const _Float16* __restrict__ Vh,
                                                  const float* __restrict__ gamma,
                                                  float* __restrict__ out) {
  const int b = blockIdx.y;
  const int t = threadIdx.x;
  const int wv = t >> 5, lane = t & 31, h = lane >> 4, r = lane & 15;
  const int qb = blockIdx.x * 128 + wv * 16;

  __shared__ float    Sl[8][16][33];             // wave-private score tile [q][32 keys]
  __shared__ _Float16 Pl[8][16][32];             // softmaxed probs (f16)
  __shared__ float    mst[8][16], lst[8][16], cst[8][16];

  if (lane < 16) { mst[wv][lane] = -1e30f; lst[wv][lane] = 0.f; }

  F16Frag qf;                                    // A-frag: 16 queries x 32 dims
  {
    const _Float16* qp = Qh + ((size_t)b * N_ + qb + r) * CK_;
    qf.h[0] = *(const v8h*)(qp + h * 8);
    qf.h[1] = *(const v8h*)(qp + 16 + h * 8);
  }
  v8f zero;
#pragma unroll
  for (int j = 0; j < 8; ++j) zero[j] = 0.f;
  v8f acc[16];
#pragma unroll
  for (int ct = 0; ct < 16; ++ct) acc[ct] = zero;

  for (int kb = 0; kb < N_; kb += 32) {
    if (kb + 32 < N_)                            // global_prefetch_b8 next K block
      __builtin_prefetch(Kh + ((size_t)b * N_ + kb + 32 + r) * CK_, 0, 0);

    // B-frags: lane = key column kb(+16)+r, dims h*8..+7 and 16+h*8..+7 per lane
    F16Frag kf0, kf1;
    const _Float16* kp0 = Kh + ((size_t)b * N_ + kb + r) * CK_ + h * 8;
    const _Float16* kp1 = kp0 + 16 * CK_;
    kf0.h[0] = *(const v8h*)kp0; kf0.h[1] = *(const v8h*)(kp0 + 16);
    kf1.h[0] = *(const v8h*)kp1; kf1.h[1] = *(const v8h*)(kp1 + 16);

    v8f s0 = __builtin_amdgcn_wmma_f32_16x16x32_f16(false, qf.v, false, kf0.v,
                                                    (short)0, zero, false, false);
    v8f s1 = __builtin_amdgcn_wmma_f32_16x16x32_f16(false, qf.v, false, kf1.v,
                                                    (short)0, zero, false, false);
#pragma unroll
    for (int j = 0; j < 8; ++j) {                // D: row q=j+8h, col key=r
      Sl[wv][8 * h + j][r]      = s0[j];
      Sl[wv][8 * h + j][16 + r] = s1[j];
    }
    asm volatile("s_wait_dscnt 0" ::: "memory"); // intra-wave LDS RAW

    if (lane < 16) {                             // one lane per query row
      float* srow = &Sl[wv][lane][0];
      float mx = srow[0];
#pragma unroll
      for (int jj = 1; jj < 32; ++jj) mx = fmaxf(mx, srow[jj]);
      float mo = mst[wv][lane];
      float mn = fmaxf(mo, mx);
      float corr = __expf(mo - mn);              // first iter: exp(-1e30)=0
      float sum = 0.f;
#pragma unroll
      for (int jj = 0; jj < 32; ++jj) {
        float p = __expf(srow[jj] - mn);
        Pl[wv][lane][jj] = (_Float16)p;
        sum += p;
      }
      lst[wv][lane] = lst[wv][lane] * corr + sum;
      mst[wv][lane] = mn;
      cst[wv][lane] = corr;
    }
    asm volatile("s_wait_dscnt 0" ::: "memory");

#pragma unroll
    for (int j = 0; j < 8; ++j) {                // rescale O accumulators
      float cj = cst[wv][8 * h + j];
#pragma unroll
      for (int ct = 0; ct < 16; ++ct) acc[ct][j] *= cj;
    }

    F16Frag pf;                                  // reload P as A-frag
    const _Float16* pp = &Pl[wv][r][0];
    pf.h[0] = *(const v8h*)(pp + h * 8);
    pf.h[1] = *(const v8h*)(pp + 16 + h * 8);

#pragma unroll
    for (int ct = 0; ct < 16; ++ct) {            // O[16q x 16ch] += P[16x32] V[32x16]
      const _Float16* vp = Vh + ((size_t)b * C_ + ct * 16 + r) * N_ + kb + h * 8;
      F16Frag vf;
      vf.h[0] = *(const v8h*)vp;
      vf.h[1] = *(const v8h*)(vp + 16);
      acc[ct] = __builtin_amdgcn_wmma_f32_16x16x32_f16(false, pf.v, false, vf.v,
                                                       (short)0, acc[ct], false, false);
    }
  }

  const float g = gamma[0];
  float inv[8];
#pragma unroll
  for (int j = 0; j < 8; ++j) inv[j] = g / lst[wv][8 * h + j];  // l >= 1 always
#pragma unroll
  for (int ct = 0; ct < 16; ++ct)
#pragma unroll
    for (int j = 0; j < 8; ++j)
      out[((size_t)b * C_ + ct * 16 + r) * N_ + qb + 8 * h + j] = acc[ct][j] * inv[j];
}

// ---------------------------------------------------------------------------
extern "C" void kernel_launch(void* const* d_in, const int* in_sizes, int n_in,
                              void* d_out, int out_size, void* d_ws, size_t ws_size,
                              hipStream_t stream) {
  (void)in_sizes; (void)n_in; (void)out_size; (void)ws_size;
  const float* x     = (const float*)d_in[0];
  const float* Wq    = (const float*)d_in[1];
  const float* bq    = (const float*)d_in[2];
  const float* Wk    = (const float*)d_in[3];
  const float* bk    = (const float*)d_in[4];
  const float* Wv    = (const float*)d_in[5];
  const float* bv    = (const float*)d_in[6];
  const float* gamma = (const float*)d_in[7];
  float* out = (float*)d_out;

  // workspace layout (bytes): Wf 163,840 | Kh 1,048,576 | Qh 1,048,576 | Vh 8,388,608
  char* w = (char*)d_ws;
  _Float16* Wf = (_Float16*)(w);
  _Float16* Kh = (_Float16*)(w + 163840);
  _Float16* Qh = (_Float16*)(w + 163840 + 1048576);
  _Float16* Vh = (_Float16*)(w + 163840 + 2 * 1048576);

  hipLaunchKernelGGL(swizzle_w, dim3(320), dim3(256), 0, stream, Wq, Wk, Wv, Wf);
  hipLaunchKernelGGL(proj_qkv, dim3(64, 4), dim3(256), 0, stream,
                     x, Wf, bq, bk, bv, Qh, Kh, Vh);
  hipLaunchKernelGGL(flash_attn, dim3(32, 4), dim3(256), 0, stream,
                     Qh, Kh, Vh, gamma, out);
}
